// CRFLoss_63324997812773
// MI455X (gfx1250) — matrix operand order repaired and checked by reading, not compiled, not run on hardware
//
#include <hip/hip_runtime.h>

typedef __attribute__((ext_vector_type(16))) __bf16 v16bf;
typedef __attribute__((ext_vector_type(8)))  float  v8f;

#define NL 126      // num_labels
#define NS 128      // num_labels + 2  (exactly 8 WMMA tiles)
#define CHUNK 256   // steps per chunk
#define NGOLD 256   // blocks for gold-score partial reduction

union BF16Frag {
  v16bf v;
  unsigned int u[8];
};

// A 16x32 bf16 fragment: m = lane%16; lanes 0-15: v0..3 -> K 0..7, v4..7 -> K 16..23;
// lanes 16-31 get K+8.  Source is row-major [128][128] bf16 in LDS.
__device__ __forceinline__ void load_a_frags(const __bf16* Psrc, int am, int kaoff,
                                             BF16Frag Af[4]) {
#pragma unroll
  for (int kb = 0; kb < 4; ++kb)
#pragma unroll
    for (int v = 0; v < 8; ++v) {
      const int k = kb * 32 + ((v < 4) ? 2 * v : 16 + 2 * (v - 4)) + kaoff;
      Af[kb].u[v] = *(const unsigned int*)&Psrc[am * NS + k];
    }
}

// B 32x16 bf16 fragment: n = lane%16; lanes 0-15: v0..3 -> K 0..7, v4..7 -> K 16..23; lanes 16-31: K+8
// Source is TRANSPOSED [n][k] row-major bf16 in LDS so K-pairs are dword-contiguous.
__device__ __forceinline__ void load_b_frag(const __bf16* Tsrc, int n, int kb, int kboff,
                                            BF16Frag& B) {
#pragma unroll
  for (int v = 0; v < 8; ++v) {
    const int k = kb * 32 + ((v < 4) ? 2 * v : 16 + 2 * (v - 4)) + (kboff >> 1);
    B.u[v] = *(const unsigned int*)&Tsrc[n * NS + k];
  }
}

// ---------------------------------------------------------------------------
// Phase A: block k computes bf16 transfer matrix of its chunk:
//   P_k = Prod_{t in chunk} (E * diag(exp(o_t))),  max-normalized each step,
// accumulating log-scale.  Out-of-range blocks naturally emit identity/0
// (used as power-of-two padding for the tree combine).
// Only 2 barriers per step (the data-hazard minimum).
// ---------------------------------------------------------------------------
__global__ __launch_bounds__(256) void crf_chunk_kernel(
    const float* __restrict__ pred,
    const float* __restrict__ trans,
    __bf16* __restrict__ Pout,
    float* __restrict__ Sout,
    int seq_len)
{
  __shared__ __bf16 Et[NS * NS];   // Et[n*128 + k] = exp(T[k][n])  (transposed)
  __shared__ __bf16 P [NS * NS];   // running chunk product, row-major bf16
  __shared__ float  expo[NS];      // exp(o_t[j]) for the *current* step
  __shared__ float  wmax[8];       // per-wave maxima

  const int tid  = threadIdx.x;
  const int wave = tid >> 5;       // wave w owns output rows [16w, 16w+16)
  const int lane = tid & 31;
  const int nlo  = lane & 15;
  const int t0   = blockIdx.x * CHUNK;

  for (int idx = tid; idx < NS * NS; idx += 256) {
    const int i = idx >> 7, j = idx & 127;
    Et[j * NS + i] = (__bf16)__expf(trans[idx]);
    P[idx] = (i == j) ? (__bf16)1.0f : (__bf16)0.0f;   // start from identity
  }
  if (tid < NS) {                   // expo for the first step of this chunk
    const int tc = (t0 < seq_len) ? t0 : (seq_len - 1);
    const float o = (tid < NL) ? pred[(size_t)tc * NL + tid] : -1000.0f;
    expo[tid] = __expf(o);
  }
  __syncthreads();

  // Constant B fragments of E cached in registers for the whole chunk.
  BF16Frag Bf[4][8];
  {
    const int kboff = (lane < 16) ? 0 : 16;
#pragma unroll
    for (int kb = 0; kb < 4; ++kb)
#pragma unroll
      for (int nt = 0; nt < 8; ++nt)
        load_b_frag(Et, nt * 16 + nlo, kb, kboff, Bf[kb][nt]);
  }

  const int kaoff = (lane < 16) ? 0 : 8;
  const int am    = wave * 16 + nlo;
  const int rbase = (lane < 16) ? 0 : 8;   // C/D: vgpr r -> M = rbase + r
  float logscale  = 0.0f;

  for (int s = 0; s < CHUNK; ++s) {
    const int t = t0 + s;
    if (t >= seq_len) break;             // uniform across block

    // (1) A fragments of current P
    BF16Frag Af[4];
    load_a_frags(P, am, kaoff, Af);

    // (2) newP = P x E   (32 WMMAs per wave per step)
    v8f acc[8];
#pragma unroll
    for (int nt = 0; nt < 8; ++nt) {
      v8f c = {0.f, 0.f, 0.f, 0.f, 0.f, 0.f, 0.f, 0.f};
#pragma unroll
      for (int kb = 0; kb < 4; ++kb)
        c = __builtin_amdgcn_wmma_f32_16x16x32_bf16(
                false, Af[kb].v, false, Bf[kb][nt].v, (short)0, c, false, false);
      acc[nt] = c;
    }

    // (3) column scale by exp(o_t[j]); thread-local max
    float lmax = 0.0f;
#pragma unroll
    for (int nt = 0; nt < 8; ++nt) {
      const float e = expo[nt * 16 + nlo];
#pragma unroll
      for (int r = 0; r < 8; ++r) {
        const float x = acc[nt][r] * e;
        acc[nt][r] = x;
        lmax = fmaxf(lmax, x);
      }
    }

    // (4) barrier-free in-wave max, one LDS slot per wave
#pragma unroll
    for (int off = 16; off > 0; off >>= 1)
      lmax = fmaxf(lmax, __shfl_xor(lmax, off, 32));
    if (lane == 0) wmax[wave] = lmax;

    __syncthreads();   // A-reads done -> P writable; wmax visible

    // (5) global max (fixed order -> deterministic), normalize, write back P,
    //     and compute next step's expo -- all inside the same barrier interval
    float mx = wmax[0];
#pragma unroll
    for (int w = 1; w < 8; ++w) mx = fmaxf(mx, wmax[w]);
    const float inv = 1.0f / mx;
    if (tid == 0) logscale += __logf(mx);

#pragma unroll
    for (int nt = 0; nt < 8; ++nt) {
      const int col = nt * 16 + nlo;
#pragma unroll
      for (int r = 0; r < 8; ++r)
        P[(wave * 16 + rbase + r) * NS + col] = (__bf16)(acc[nt][r] * inv);
    }
    if (tid < NS) {                       // expo for step s+1 (clamped; unused on last)
      const int tn = (t + 1 < seq_len) ? t + 1 : t;
      const float o = (tid < NL) ? pred[(size_t)tn * NL + tid] : -1000.0f;
      expo[tid] = __expf(o);
    }
    __syncthreads();   // P/expo visible for next step
  }

  for (int idx = tid; idx < NS * NS; idx += 256)
    Pout[(size_t)blockIdx.x * (NS * NS) + idx] = P[idx];
  if (tid == 0) Sout[blockIdx.x] = logscale;
}

// ---------------------------------------------------------------------------
// Tree combine: block b computes  M[2b*stride] <- norm(M[2b*stride] x M[(2b+1)*stride])
// in-place with doubling stride; scales add (+ log of the new max).
// ---------------------------------------------------------------------------
__global__ __launch_bounds__(256) void crf_pair_kernel(
    __bf16* __restrict__ Pbuf, float* __restrict__ Sbuf, int stride)
{
  __shared__ __bf16 Alds[NS * NS];   // left matrix, row-major
  __shared__ __bf16 Blds[NS * NS];   // right matrix, transposed [n][k]
  __shared__ float  wmax[8];

  const int tid  = threadIdx.x;
  const int wave = tid >> 5;
  const int lane = tid & 31;
  const int nlo  = lane & 15;

  const size_t slotA = (size_t)(2 * blockIdx.x) * stride * (NS * NS);
  const size_t slotB = slotA + (size_t)stride * (NS * NS);

  // stage A (straight copy) and B (transposed) into LDS
  {
    const uint4* srcA = (const uint4*)(Pbuf + slotA);
    const uint4* srcB = (const uint4*)(Pbuf + slotB);
    uint4* dstA = (uint4*)Alds;
    for (int idx = tid; idx < (NS * NS) / 8; idx += 256) {
      dstA[idx] = srcA[idx];
      const uint4 q = srcB[idx];
      const int k = idx >> 4;              // source row
      const int nb = (idx & 15) * 8;       // source col base
      const __bf16* qh = (const __bf16*)&q;
#pragma unroll
      for (int e = 0; e < 8; ++e)
        Blds[(nb + e) * NS + k] = qh[e];
    }
  }
  __syncthreads();

  const int kaoff = (lane < 16) ? 0 : 8;
  const int kboff = (lane < 16) ? 0 : 16;
  const int am    = wave * 16 + nlo;
  const int rbase = (lane < 16) ? 0 : 8;

  BF16Frag Af[4];
  load_a_frags(Alds, am, kaoff, Af);

  v8f acc[8];
#pragma unroll
  for (int nt = 0; nt < 8; ++nt) {
    v8f c = {0.f, 0.f, 0.f, 0.f, 0.f, 0.f, 0.f, 0.f};
#pragma unroll
    for (int kb = 0; kb < 4; ++kb) {
      BF16Frag B;
      load_b_frag(Blds, nt * 16 + nlo, kb, kboff, B);
      c = __builtin_amdgcn_wmma_f32_16x16x32_bf16(
              false, Af[kb].v, false, B.v, (short)0, c, false, false);
    }
    acc[nt] = c;
  }

  float lmax = 0.0f;
#pragma unroll
  for (int nt = 0; nt < 8; ++nt)
#pragma unroll
    for (int r = 0; r < 8; ++r) lmax = fmaxf(lmax, acc[nt][r]);
#pragma unroll
  for (int off = 16; off > 0; off >>= 1)
    lmax = fmaxf(lmax, __shfl_xor(lmax, off, 32));
  if (lane == 0) wmax[wave] = lmax;
  __syncthreads();

  float mx = wmax[0];
#pragma unroll
  for (int w = 1; w < 8; ++w) mx = fmaxf(mx, wmax[w]);
  const float inv = 1.0f / mx;

#pragma unroll
  for (int nt = 0; nt < 8; ++nt) {
    const int col = nt * 16 + nlo;
#pragma unroll
    for (int r = 0; r < 8; ++r)
      Pbuf[slotA + (wave * 16 + rbase + r) * NS + col] = (__bf16)(acc[nt][r] * inv);
  }
  if (tid == 0) {
    const size_t sA = (size_t)(2 * blockIdx.x) * stride;
    Sbuf[sA] = Sbuf[sA] + Sbuf[sA + stride] + __logf(mx);
  }
}

// ---------------------------------------------------------------------------
// Gold path score partials (fixed-order tree -> deterministic)
// ---------------------------------------------------------------------------
__global__ __launch_bounds__(256) void gold_partial_kernel(
    const float* __restrict__ pred, const float* __restrict__ trans,
    const int* __restrict__ ref, float* __restrict__ partial, int seq_len)
{
  __shared__ float red[256];
  const int tid = threadIdx.x;
  const int gid = blockIdx.x * 256 + tid;
  const int stride = gridDim.x * 256;
  float s = 0.0f;
  for (int t = gid; t < seq_len; t += stride) {
    const int r = ref[t];
    s += pred[(size_t)t * NL + r];
    if (t + 1 < seq_len) s += trans[r * NS + ref[t + 1]];
  }
  red[tid] = s;
  __syncthreads();
  for (int off = 128; off > 0; off >>= 1) {
    if (tid < off) red[tid] += red[tid + off];
    __syncthreads();
  }
  if (tid == 0) partial[blockIdx.x] = red[0];
}

// ---------------------------------------------------------------------------
// Final: start vector is one-hot e_126, so all-paths score = row 126 of the
// tree-combined matrix dotted with exp(T[:,127]), plus accumulated log-scale.
// ---------------------------------------------------------------------------
__global__ __launch_bounds__(256) void crf_final_kernel(
    const __bf16* __restrict__ Pfinal, const float* __restrict__ Sbuf,
    const float* __restrict__ trans, const int* __restrict__ ref,
    const float* __restrict__ goldPartial, int seq_len, float* __restrict__ out)
{
  __shared__ float red[256];
  const int tid = threadIdx.x;
  float term = 0.0f;
  if (tid < NS)
    term = (float)Pfinal[NL * NS + tid] * __expf(trans[tid * NS + (NS - 1)]);
  red[tid] = term;
  __syncthreads();
  for (int off = 128; off > 0; off >>= 1) {
    if (tid < off) red[tid] += red[tid + off];
    __syncthreads();
  }
  if (tid == 0) {
    const float allp = __logf(red[0]) + Sbuf[0];
    float gold = trans[NL * NS + ref[0]] +
                 trans[(size_t)ref[seq_len - 1] * NS + (NS - 1)];
    for (int b = 0; b < NGOLD; ++b) gold += goldPartial[b];
    out[0] = allp - gold;
  }
}

// ---------------------------------------------------------------------------
extern "C" void kernel_launch(void* const* d_in, const int* in_sizes, int n_in,
                              void* d_out, int out_size, void* d_ws, size_t ws_size,
                              hipStream_t stream) {
  const float* pred  = (const float*)d_in[0];   // [seq_len, 126] f32
  const float* trans = (const float*)d_in[1];   // [128, 128] f32
  const int*   ref   = (const int*)d_in[2];     // [seq_len] i32
  float* out = (float*)d_out;

  const int seq_len = in_sizes[2];
  const int nc = (seq_len + CHUNK - 1) / CHUNK;
  int ncp = 1; while (ncp < nc) ncp <<= 1;      // pad to power of two (512 here)

  char* ws = (char*)d_ws;
  __bf16* Pbuf = (__bf16*)ws;                                     // ncp * 32KB
  const size_t psz = (size_t)ncp * NS * NS * sizeof(__bf16);
  float* Sbuf = (float*)(ws + psz);                               // ncp floats
  float* goldPartial = Sbuf + ncp;                                // NGOLD floats

  crf_chunk_kernel<<<ncp, 256, 0, stream>>>(pred, trans, Pbuf, Sbuf, seq_len);
  gold_partial_kernel<<<NGOLD, 256, 0, stream>>>(pred, trans, ref, goldPartial, seq_len);

  int stride = 1, count = ncp;
  while (count > 1) {                            // 9 levels for 512 chunks
    crf_pair_kernel<<<count >> 1, 256, 0, stream>>>(Pbuf, Sbuf, stride);
    stride <<= 1; count >>= 1;
  }
  crf_final_kernel<<<1, 256, 0, stream>>>(Pbuf, Sbuf, trans, ref, goldPartial,
                                          seq_len, out);
}
